// AnomalyTransformer_48421461296007
// MI455X (gfx1250) — matrix operand order repaired and checked by reading, not compiled, not run on hardware
//
#include <hip/hip_runtime.h>
#include <hip/hip_bf16.h>

typedef __attribute__((ext_vector_type(16))) _Float16 v16h;
typedef __attribute__((ext_vector_type(8)))  _Float16 v8h;
typedef __attribute__((ext_vector_type(8)))  float    v8f;

#define D_MODEL 512
#define N_HEADS 8
#define EHEAD   64
#define WIN     512

// GEMM tiling: 256 threads = 8 waves (wave32); block tile 128x64; wave tile 32x32 (2x2 WMMA)
#define BM 128
#define BN 64
#define BK 32
#define PAD 8

enum { MODE_STD = 0, MODE_QK = 1, MODE_VT = 2, MODE_ATTN = 3 };
enum { ACT_NONE = 0, ACT_GELU = 1 };

// ---- CDNA5 async global->LDS path (guarded so compile stays clean either way) ----
#if defined(__gfx1250__) && __has_builtin(__builtin_amdgcn_global_load_async_to_lds_b128) && \
    __has_builtin(__builtin_amdgcn_s_wait_asynccnt)
#define ASYNC_LDS 1
typedef int a_v4i __attribute__((__vector_size__(4 * sizeof(int))));
typedef __attribute__((address_space(1))) a_v4i* g4p;   // global (AS1) v4i*
typedef __attribute__((address_space(3))) a_v4i* l4p;   // LDS (AS3) v4i*
__device__ __forceinline__ void async_cp16(const void* g, void* l) {
  __builtin_amdgcn_global_load_async_to_lds_b128((g4p)g, (l4p)l, 0, 0);
}
#define WAIT_ASYNC() __builtin_amdgcn_s_wait_asynccnt(0)
#else
#define WAIT_ASYNC()
#endif

// ---------------------------------------------------------------------------
// Generic f16 WMMA GEMM: C = A[MxK] * BT[NxK]^T + bias, batched via blockIdx.z
// Double-buffered LDS; async copy (or software-pipelined reg staging) overlaps
// the next K tile's global traffic with the current tile's WMMA work.
// ---------------------------------------------------------------------------
__global__ __launch_bounds__(256) void k_gemm(
    const _Float16* __restrict__ A, unsigned long long strideA,
    const _Float16* __restrict__ BT, unsigned long long strideB,
    const float* __restrict__ bias,
    float* __restrict__ C32, _Float16* __restrict__ C16,
    int M, int N, int K, int act, int mode)
{
  __shared__ __align__(16) _Float16 As[2][BM][BK + PAD];
  __shared__ __align__(16) _Float16 Bs[2][BN][BK + PAD];

  const int tid  = threadIdx.x;
  const int lane = tid & 31;
  const int wv   = tid >> 5;
  const int wm   = wv >> 1;          // 0..3
  const int wn   = wv & 1;           // 0..1
  const int bm0  = blockIdx.y * BM;
  const int bn0  = blockIdx.x * BN;
  const int z    = blockIdx.z;

  const _Float16* Ab = A  + (unsigned long long)z * strideA;
  const _Float16* Bb = BT + (unsigned long long)z * strideB;

  // per-thread staging addresses: 2 x 16B for A (rows r, r+64), 1 x 16B for B
  const int srow = tid >> 2;                 // 0..63
  const int sc8  = (tid & 3) << 3;           // 0,8,16,24 halves
  const _Float16* gA0 = Ab + (unsigned long long)(bm0 + srow) * (unsigned long long)K + sc8;
  const _Float16* gA1 = gA0 + 64ULL * (unsigned long long)K;
  const _Float16* gB0 = Bb + (unsigned long long)(bn0 + srow) * (unsigned long long)K + sc8;

  uint4 ra0, ra1, rb0;                       // used only by the sync fallback
  auto stage_issue = [&](int buf, int kc) {
#ifdef ASYNC_LDS
    async_cp16(gA0 + kc, &As[buf][srow][sc8]);
    async_cp16(gA1 + kc, &As[buf][64 + srow][sc8]);
    async_cp16(gB0 + kc, &Bs[buf][srow][sc8]);
#else
    ra0 = *(const uint4*)(gA0 + kc);
    ra1 = *(const uint4*)(gA1 + kc);
    rb0 = *(const uint4*)(gB0 + kc);
#endif
  };
  auto stage_commit = [&](int buf) {
#ifndef ASYNC_LDS
    *(uint4*)(&As[buf][srow][sc8])      = ra0;
    *(uint4*)(&As[buf][64 + srow][sc8]) = ra1;
    *(uint4*)(&Bs[buf][srow][sc8])      = rb0;
#endif
    (void)buf;
  };

  v8f zero = {0.f, 0.f, 0.f, 0.f, 0.f, 0.f, 0.f, 0.f};
  v8f acc[2][2];
  acc[0][0] = zero; acc[0][1] = zero; acc[1][0] = zero; acc[1][1] = zero;

  const int mrow = lane & 15;
  const int khA  = (lane < 16) ? 0 : 8;    // A: K half-groups per ISA 7.12.2
  const int kbB  = (lane < 16) ? 0 : 8;    // B: K half-groups matching A
  const int nKb  = K / BK;

  // prologue: tile 0 into buffer 0
  stage_issue(0, 0);
  stage_commit(0);

  for (int kb = 0; kb < nKb; ++kb) {
    WAIT_ASYNC();              // this wave's async writes to LDS are complete
    __syncthreads();           // every wave's writes are complete
    const int buf = kb & 1;
    const bool hasNext = (kb + 1) < nKb;
    if (hasNext) stage_issue(buf ^ 1, (kb + 1) * BK);   // overlap with WMMA below

    v16h af[2], bf[2];
#pragma unroll
    for (int tm = 0; tm < 2; ++tm) {
      const _Float16* p = &As[buf][wm * 32 + tm * 16 + mrow][0];
      v8h lo = *(const v8h*)(p + khA);        // K = khA .. khA+7
      v8h hi = *(const v8h*)(p + khA + 16);   // K = khA+16 .. khA+23
#pragma unroll
      for (int i = 0; i < 8; ++i) { af[tm][i] = lo[i]; af[tm][i + 8] = hi[i]; }
    }
#pragma unroll
    for (int tn = 0; tn < 2; ++tn) {
      const _Float16* p = &Bs[buf][wn * 32 + tn * 16 + mrow][0];
      v8h lo = *(const v8h*)(p + kbB);        // K = kbB .. kbB+7
      v8h hi = *(const v8h*)(p + kbB + 16);   // K = kbB+16 .. kbB+23
#pragma unroll
      for (int i = 0; i < 8; ++i) { bf[tn][i] = lo[i]; bf[tn][i + 8] = hi[i]; }
    }
#pragma unroll
    for (int tm = 0; tm < 2; ++tm)
#pragma unroll
      for (int tn = 0; tn < 2; ++tn)
        acc[tm][tn] = __builtin_amdgcn_wmma_f32_16x16x32_f16(
            false, af[tm], false, bf[tn], (short)0, acc[tm][tn], false, false);

    if (hasNext) stage_commit(buf ^ 1);       // sync path: ds_store after compute
  }

  // ---- epilogue: C layout — VGPR r: lanes 0-15 -> M=r, lanes 16-31 -> M=8+r ----
  const int rowAdd = (lane < 16) ? 0 : 8;
#pragma unroll
  for (int tm = 0; tm < 2; ++tm) {
#pragma unroll
    for (int tn = 0; tn < 2; ++tn) {
      const int gn = bn0 + wn * 32 + tn * 16 + mrow;
      const float bvv = bias ? bias[gn] : 0.0f;
#pragma unroll
      for (int r = 0; r < 8; ++r) {
        const int gm = bm0 + wm * 32 + tm * 16 + rowAdd + r;
        float v = acc[tm][tn][r] + bvv;
        if (act == ACT_GELU) v = 0.5f * v * (1.0f + erff(v * 0.70710678118654752f));
        if (mode == MODE_STD) {
          unsigned long long idx = (unsigned long long)z * (unsigned long long)M * (unsigned long long)N
                                 + (unsigned long long)gm * (unsigned long long)N + gn;
          if (C32) C32[idx] = v;
          if (C16) C16[idx] = (_Float16)v;
        } else if (mode == MODE_QK) {        // m = b*512+l, n = h*64+e  ->  [b,h,l,e]
          int b = gm >> 9, l = gm & 511;
          int hh = gn >> 6, e = gn & 63;
          C16[(((unsigned long long)(b * N_HEADS + hh) * WIN + l) * EHEAD) + e] = (_Float16)v;
        } else if (mode == MODE_VT) {        // -> [b,h,e,l]
          int b = gm >> 9, l = gm & 511;
          int hh = gn >> 6, e = gn & 63;
          C16[(((unsigned long long)(b * N_HEADS + hh) * EHEAD + e) * WIN) + l] = (_Float16)v;
        } else {                              // MODE_ATTN: z=b*8+h, out[b,l,h*64+e]
          int b = z >> 3, hh = z & 7;
          unsigned long long idx = ((unsigned long long)b * WIN + gm) * D_MODEL + hh * EHEAD + gn;
          if (C32) C32[idx] = v;
          if (C16) C16[idx] = (_Float16)v;
        }
      }
    }
  }
}

// ---------------------------------------------------------------------------
// Weight transpose + fp16 convert: W[512x512] f32 -> WT[512x512] f16 (N-major)
// ---------------------------------------------------------------------------
__global__ __launch_bounds__(256) void k_wtrans(const float* __restrict__ W,
                                                _Float16* __restrict__ WT)
{
  unsigned t = blockIdx.x * 256u + threadIdx.x;   // 512*512 / 256 = 1024 blocks
  unsigned n = t & 511u, k = t >> 9;
  WT[(unsigned long long)n * D_MODEL + k] = (_Float16)W[(unsigned long long)k * D_MODEL + n];
}

// ---------------------------------------------------------------------------
// DataEmbedding: wrap-pad conv1d(k=3) over C_IN=38 + sinusoidal PE
// ---------------------------------------------------------------------------
__global__ __launch_bounds__(256) void k_embed(const float* __restrict__ x,
                                               const float* __restrict__ ck,
                                               float* __restrict__ h32,
                                               _Float16* __restrict__ h16)
{
  __shared__ float xs[3 * 38];
  const int bw = blockIdx.x;            // b*512 + w
  const int b = bw >> 9, w = bw & 511;
  const int tid = threadIdx.x;
  if (tid < 114) {
    int r = tid / 38, c = tid - r * 38;
    int src = w - 1 + r;
    src = (src < 0) ? (WIN - 1) : ((src >= WIN) ? 0 : src);
    xs[tid] = x[((unsigned long long)b * WIN + src) * 38 + c];
  }
  __syncthreads();
#pragma unroll
  for (int t = 0; t < 2; ++t) {
    int d = tid + t * 256;
    float acc = 0.0f;
    for (int rc = 0; rc < 114; ++rc)
      acc += xs[rc] * ck[(unsigned long long)rc * D_MODEL + d];
    int j2 = d & ~1;
    float divv = expf(-(float)j2 * (9.210340371976184f / 512.0f));
    float ang  = (float)w * divv;
    acc += (d & 1) ? cosf(ang) : sinf(ang);
    unsigned long long idx = (unsigned long long)bw * D_MODEL + d;
    h32[idx] = acc;
    h16[idx] = (_Float16)acc;
  }
}

// ---------------------------------------------------------------------------
// Sigma head: sg[b,h,l] = 3^(sigmoid(5*(h.Ws+bs))+1e-5) - 1
// ---------------------------------------------------------------------------
__global__ __launch_bounds__(256) void k_sigma(const float* __restrict__ h32,
                                               const float* __restrict__ Ws,
                                               const float* __restrict__ bs,
                                               float* __restrict__ sg)
{
  __shared__ float hrow[D_MODEL];
  const int row = blockIdx.x;           // b*512 + l
  const int tid = threadIdx.x;
  hrow[tid]       = h32[(unsigned long long)row * D_MODEL + tid];
  hrow[tid + 256] = h32[(unsigned long long)row * D_MODEL + tid + 256];
  __syncthreads();
  const int lane = tid & 31, hh = tid >> 5;   // one head per wave32
  float s = 0.0f;
  for (int j = lane; j < D_MODEL; j += 32)
    s += hrow[j] * Ws[j * N_HEADS + hh];
#pragma unroll
  for (int o = 16; o > 0; o >>= 1) s += __shfl_xor(s, o, 32);
  if (lane == 0) {
    float v = s + bs[hh];
    float sig = 1.0f / (1.0f + expf(-5.0f * v)) + 1e-5f;
    float g = powf(3.0f, sig) - 1.0f;
    int b = row >> 9, l = row & 511;
    sg[(unsigned long long)(b * N_HEADS + hh) * WIN + l] = g;
  }
}

// ---------------------------------------------------------------------------
// Row softmax of scale*scores; writes series f32 (d_out) and f16 (for series@V)
// ---------------------------------------------------------------------------
__global__ __launch_bounds__(256) void k_softmax(const float* __restrict__ scores,
                                                 float* __restrict__ series32,
                                                 _Float16* __restrict__ series16)
{
  __shared__ float red[256];
  const unsigned long long base = (unsigned long long)blockIdx.x * WIN;
  const int tid = threadIdx.x;
  float x0 = scores[base + tid] * 0.125f;
  float x1 = scores[base + tid + 256] * 0.125f;
  float m = fmaxf(x0, x1);
  red[tid] = m; __syncthreads();
  for (int o = 128; o > 0; o >>= 1) { if (tid < o) red[tid] = fmaxf(red[tid], red[tid + o]); __syncthreads(); }
  m = red[0]; __syncthreads();
  float e0 = expf(x0 - m), e1 = expf(x1 - m);
  red[tid] = e0 + e1; __syncthreads();
  for (int o = 128; o > 0; o >>= 1) { if (tid < o) red[tid] += red[tid + o]; __syncthreads(); }
  float inv = 1.0f / red[0];
  float y0 = e0 * inv, y1 = e1 * inv;
  series32[base + tid]       = y0;
  series32[base + tid + 256] = y1;
  series16[base + tid]       = (_Float16)y0;
  series16[base + tid + 256] = (_Float16)y1;
}

// ---------------------------------------------------------------------------
// Gaussian prior + sigma broadcast (pure streaming writes, HBM-bound)
// ---------------------------------------------------------------------------
__global__ __launch_bounds__(256) void k_prior(const float* __restrict__ sg,
                                               float* __restrict__ prior,
                                               float* __restrict__ sigma)
{
  const unsigned bhl = blockIdx.x;      // (b*8+h)*512 + l
  const int l = bhl & 511;
  const float g = sg[bhl];
  const unsigned long long base = (unsigned long long)bhl * WIN;
  const float c = 0.3989422804014327f / g;
  const float e = -1.0f / (2.0f * g * g);
  const int tid = threadIdx.x;
#pragma unroll
  for (int t = 0; t < 2; ++t) {
    int s = tid + t * 256;
    float d = (float)(l - s);
    prior[base + s] = c * expf(d * d * e);
    sigma[base + s] = g;
  }
}

// ---------------------------------------------------------------------------
// out = LayerNorm(a [+ b]) * gamma + beta  (eps=1e-3); emits f32 and f16
// ---------------------------------------------------------------------------
__global__ __launch_bounds__(256) void k_add_ln(const float* __restrict__ a,
                                                const float* __restrict__ bsrc,
                                                const float* __restrict__ gamma,
                                                const float* __restrict__ beta,
                                                float* __restrict__ o32,
                                                _Float16* __restrict__ o16)
{
  __shared__ float red[256];
  __shared__ float stat0;
  const unsigned long long base = (unsigned long long)blockIdx.x * D_MODEL;
  const int tid = threadIdx.x;
  float x0 = a[base + tid], x1 = a[base + tid + 256];
  if (bsrc) { x0 += bsrc[base + tid]; x1 += bsrc[base + tid + 256]; }
  red[tid] = x0 + x1; __syncthreads();
  for (int o = 128; o > 0; o >>= 1) { if (tid < o) red[tid] += red[tid + o]; __syncthreads(); }
  if (tid == 0) stat0 = red[0] * (1.0f / 512.0f);
  __syncthreads();
  float mu = stat0;
  float d0 = x0 - mu, d1 = x1 - mu;
  red[tid] = d0 * d0 + d1 * d1; __syncthreads();
  for (int o = 128; o > 0; o >>= 1) { if (tid < o) red[tid] += red[tid + o]; __syncthreads(); }
  float var = red[0] * (1.0f / 512.0f);
  float rstd = rsqrtf(var + 1e-3f);
  float y0 = d0 * rstd * gamma[tid] + beta[tid];
  float y1 = d1 * rstd * gamma[tid + 256] + beta[tid + 256];
  o32[base + tid] = y0; o32[base + tid + 256] = y1;
  if (o16) { o16[base + tid] = (_Float16)y0; o16[base + tid + 256] = (_Float16)y1; }
}

// ---------------------------------------------------------------------------
// Final projection: out[b,l,c] = h . Wp[:,c] + bp[c]   (C_OUT = 38)
// ---------------------------------------------------------------------------
__global__ __launch_bounds__(256) void k_proj(const float* __restrict__ h32,
                                              const float* __restrict__ Wp,
                                              const float* __restrict__ bp,
                                              float* __restrict__ out)
{
  __shared__ float hrow[D_MODEL];
  const int row = blockIdx.x;
  const int tid = threadIdx.x;
  hrow[tid]       = h32[(unsigned long long)row * D_MODEL + tid];
  hrow[tid + 256] = h32[(unsigned long long)row * D_MODEL + tid + 256];
  __syncthreads();
  const int lane = tid & 31, wv = tid >> 5;
  for (int c = wv; c < 38; c += 8) {
    float s = 0.0f;
    for (int j = lane; j < D_MODEL; j += 32)
      s += hrow[j] * Wp[j * 38 + c];
#pragma unroll
    for (int o = 16; o > 0; o >>= 1) s += __shfl_xor(s, o, 32);
    if (lane == 0) out[(unsigned long long)row * 38 + c] = s + bp[c];
  }
}

// ---------------------------------------------------------------------------
extern "C" void kernel_launch(void* const* d_in, const int* in_sizes, int n_in,
                              void* d_out, int out_size, void* d_ws, size_t ws_size,
                              hipStream_t stream)
{
  (void)in_sizes; (void)n_in; (void)out_size; (void)ws_size;
  const float* x      = (const float*)d_in[0];
  const float* conv_k = (const float*)d_in[1];
  const float* Wq = (const float*)d_in[2];  const float* bq = (const float*)d_in[3];
  const float* Wk = (const float*)d_in[4];  const float* bk = (const float*)d_in[5];
  const float* Wv = (const float*)d_in[6];  const float* bv = (const float*)d_in[7];
  const float* Ws = (const float*)d_in[8];  const float* bs = (const float*)d_in[9];
  const float* Wo = (const float*)d_in[10]; const float* bo = (const float*)d_in[11];
  const float* W1 = (const float*)d_in[12]; const float* b1 = (const float*)d_in[13];
  const float* W2 = (const float*)d_in[14]; const float* b2 = (const float*)d_in[15];
  const float* ln1_g = (const float*)d_in[16]; const float* ln1_b = (const float*)d_in[17];
  const float* ln2_g = (const float*)d_in[18]; const float* ln2_b = (const float*)d_in[19];
  const float* lnf_g = (const float*)d_in[20]; const float* lnf_b = (const float*)d_in[21];
  const float* Wp = (const float*)d_in[22]; const float* bp = (const float*)d_in[23];

  float* out = (float*)d_out;
  const unsigned long long OUT_N = 8ULL * 512 * 38;            // 155648
  const unsigned long long MAPL  = 64ULL * 512 * 512;          // per-layer map size
  float* series_out = out + OUT_N;
  float* prior_out  = series_out + 3ULL * MAPL;
  float* sigma_out  = prior_out  + 3ULL * MAPL;

  // ---- workspace carve-out (d_ws is 256B aligned) ----
  char* wbase = (char*)d_ws;
  size_t off = 0;
  auto alloc = [&](size_t bytes) -> void* {
    void* p = wbase + off;
    off = (off + bytes + 255) & ~(size_t)255;
    return p;
  };
  float*    h32    = (float*)   alloc(4096ULL * 512 * 4);
  _Float16* h16    = (_Float16*)alloc(4096ULL * 512 * 2);
  _Float16* q16    = (_Float16*)alloc(64ULL * 512 * 64 * 2);
  _Float16* k16    = (_Float16*)alloc(64ULL * 512 * 64 * 2);
  _Float16* v16    = (_Float16*)alloc(64ULL * 64 * 512 * 2);
  float*    sg     = (float*)   alloc(64ULL * 512 * 4);
  float*    scores = (float*)   alloc(64ULL * 512 * 512 * 4);
  _Float16* ser16  = (_Float16*)alloc(64ULL * 512 * 512 * 2);
  float*    attn32 = (float*)   alloc(4096ULL * 512 * 4);
  _Float16* attn16 = (_Float16*)alloc(4096ULL * 512 * 2);
  _Float16* y16    = (_Float16*)alloc(4096ULL * 512 * 2);
  _Float16* wT     = (_Float16*)alloc(18ULL * 512 * 512 * 2);
  // o32 / z32 reuse the scores region (scores are consumed by softmax first)
  float* o32 = scores;
  float* z32 = scores + 4096ULL * 512;

  const unsigned long long DD = 512ULL * 512;

  // ---- weights -> transposed fp16 (B operand [N x K]) ----
  for (int i = 0; i < 3; ++i) {
    const float* srcs[6] = { Wq + i * DD, Wk + i * DD, Wv + i * DD,
                             Wo + i * DD, W1 + i * DD, W2 + i * DD };
    for (int j = 0; j < 6; ++j)
      k_wtrans<<<1024, 256, 0, stream>>>(srcs[j], wT + (unsigned long long)(i * 6 + j) * DD);
  }

  k_embed<<<4096, 256, 0, stream>>>(x, conv_k, h32, h16);

  dim3 gBig(512 / BN, 4096 / BM, 1);    // projections / FFN
  dim3 gScore(512 / BN, 512 / BM, 64);  // q.k^T batched per (b,h)
  dim3 gAV(64 / BN, 512 / BM, 64);      // series @ v batched per (b,h)

  for (int i = 0; i < 3; ++i) {
    _Float16* WqT = wT + (unsigned long long)(i * 6 + 0) * DD;
    _Float16* WkT = wT + (unsigned long long)(i * 6 + 1) * DD;
    _Float16* WvT = wT + (unsigned long long)(i * 6 + 2) * DD;
    _Float16* WoT = wT + (unsigned long long)(i * 6 + 3) * DD;
    _Float16* W1T = wT + (unsigned long long)(i * 6 + 4) * DD;
    _Float16* W2T = wT + (unsigned long long)(i * 6 + 5) * DD;

    // Q/K/V projections with per-head output layouts
    k_gemm<<<gBig, 256, 0, stream>>>(h16, 0ULL, WqT, 0ULL, bq + i * 512,
                                     nullptr, q16, 4096, 512, 512, ACT_NONE, MODE_QK);
    k_gemm<<<gBig, 256, 0, stream>>>(h16, 0ULL, WkT, 0ULL, bk + i * 512,
                                     nullptr, k16, 4096, 512, 512, ACT_NONE, MODE_QK);
    k_gemm<<<gBig, 256, 0, stream>>>(h16, 0ULL, WvT, 0ULL, bv + i * 512,
                                     nullptr, v16, 4096, 512, 512, ACT_NONE, MODE_VT);
    k_sigma<<<4096, 256, 0, stream>>>(h32, Ws + i * 512 * N_HEADS, bs + i * N_HEADS, sg);

    // scores = q.k^T, softmax -> series; prior/sigma streaming writes
    k_gemm<<<gScore, 256, 0, stream>>>(q16, 512ULL * 64, k16, 512ULL * 64, nullptr,
                                       scores, nullptr, 512, 512, 64, ACT_NONE, MODE_STD);
    k_softmax<<<64 * 512, 256, 0, stream>>>(scores, series_out + (unsigned long long)i * MAPL, ser16);
    k_prior<<<64 * 512, 256, 0, stream>>>(sg, prior_out + (unsigned long long)i * MAPL,
                                          sigma_out + (unsigned long long)i * MAPL);

    // attn = series @ v ; project with Wo ; residual + LN1
    k_gemm<<<gAV, 256, 0, stream>>>(ser16, 512ULL * 512, v16, 64ULL * 512, nullptr,
                                    attn32, attn16, 512, 64, 512, ACT_NONE, MODE_ATTN);
    k_gemm<<<gBig, 256, 0, stream>>>(attn16, 0ULL, WoT, 0ULL, bo + i * 512,
                                     o32, nullptr, 4096, 512, 512, ACT_NONE, MODE_STD);
    k_add_ln<<<4096, 256, 0, stream>>>(h32, o32, ln1_g + i * 512, ln1_b + i * 512, h32, h16);

    // FFN: GELU(h W1 + b1) W2 + b2 ; residual + LN2
    k_gemm<<<gBig, 256, 0, stream>>>(h16, 0ULL, W1T, 0ULL, b1 + i * 512,
                                     nullptr, y16, 4096, 512, 512, ACT_GELU, MODE_STD);
    k_gemm<<<gBig, 256, 0, stream>>>(y16, 0ULL, W2T, 0ULL, b2 + i * 512,
                                     z32, nullptr, 4096, 512, 512, ACT_NONE, MODE_STD);
    k_add_ln<<<4096, 256, 0, stream>>>(h32, z32, ln2_g + i * 512, ln2_b + i * 512, h32, h16);
  }

  // final LN + 38-wide projection
  k_add_ln<<<4096, 256, 0, stream>>>(h32, nullptr, lnf_g, lnf_b, h32, nullptr);
  k_proj<<<4096, 256, 0, stream>>>(h32, Wp, bp, out);
}
